// PCNO_56813827392079
// MI455X (gfx1250) — matrix operand order repaired and checked by reading, not compiled, not run on hardware
//
#include <hip/hip_runtime.h>
#include <math.h>

// Problem constants (match reference)
#define BB 2
#define NN 16000
#define EE 160000
#define CC 64
#define KK 128
#define LL 3

typedef __attribute__((ext_vector_type(16))) __bf16 v16bf;
typedef __attribute__((ext_vector_type(8)))  float  v8f;

__device__ __forceinline__ float gelu_tanh(float x) {
    float x3 = x * x * x;
    return 0.5f * x * (1.0f + tanhf(0.7978845608028654f * (x + 0.044715f * x3)));
}

// ---- fragment-order LDS addressing (inverse of ISA 7.12.2 layouts) ----
// A 16x32 bf16 tile: lane = (row&15) | (((k>>3)&1)<<4); elem = (k&7) | ((k>>4)<<3)
__device__ __forceinline__ int a_idx(int row, int k) {
    int lane = (row & 15) | (((k >> 3) & 1) << 4);
    return (((row >> 4) << 5) + lane) * 16 + ((k & 7) | ((k >> 4) << 3));
}
// B 32x16 bf16 tile: lane = (col&15) | (((k>>3)&1)<<4); elem = (k&7) | ((k>>4)<<3)
__device__ __forceinline__ int b_idx(int k, int col) {
    int lane = (col & 15) | (((k >> 3) & 1) << 4);
    return (((col >> 4) << 5) + lane) * 16 + ((k & 7) | ((k >> 4) << 3));
}

union BF2 { unsigned u; __bf16 e[2]; };
__device__ __forceinline__ unsigned pack_bf2(float a, float b) {
    BF2 r; r.e[0] = (__bf16)a; r.e[1] = (__bf16)b; return r.u;
}

// fragment load: 16 contiguous bf16 per lane -> 2x ds_load_b128
__device__ __forceinline__ v16bf frag_ld(const __bf16* p) {
    union { uint4 u[2]; v16bf v; } t;
    const uint4* q = (const uint4*)p;
    t.u[0] = q[0];
    t.u[1] = q[1];
    return t.v;
}

#define WMMA_BF16(a, b, c) \
    __builtin_amdgcn_wmma_f32_16x16x32_bf16(false, (a), false, (b), (short)0, (c), false, false)

// stage a 64/128-row x 32-col bf16 A panel (row-major source) into frag order
__device__ __forceinline__ void stage_a_bf16(__bf16* As, const __bf16* src,
                                             int srcStride, int t, int nIter) {
#pragma unroll
    for (int jj = 0; jj < 8; ++jj) {
        if (jj >= nIter) break;
        int p = t + 256 * jj;
        int row = p >> 4, k = (p & 15) << 1;
        unsigned v = *(const unsigned*)(src + row * srcStride + k);
        *(unsigned*)(As + a_idx(row, k)) = v;
    }
}

// ------------------------------------------------------------------
__global__ void k_zero4(float4* p, int n4) {
    int i = blockIdx.x * 256 + threadIdx.x;
    if (i < n4) p[i] = make_float4(0.f, 0.f, 0.f, 0.f);
}

__global__ void k_prep(const float* ws_W, const float* gws_W, const float* fc1_W,
                       __bf16* wsb, __bf16* gwsb, __bf16* fc1b) {
    int i = blockIdx.x * 256 + threadIdx.x;
    const int n1 = LL * CC * CC, n2 = LL * CC * KK, n3 = KK * CC;
    if (i < n1) wsb[i] = (__bf16)ws_W[i];
    else if (i < n1 + n2) gwsb[i - n1] = (__bf16)gws_W[i - n1];
    else if (i < n1 + n2 + n3) fc1b[i - n1 - n2] = (__bf16)fc1_W[i - n1 - n2];
}

// bases: Bc/Bs bf16 stored [b][k][n]
__global__ void k_bases(const float* nodes, const float* modes, const float* latent,
                        __bf16* Bc, __bf16* Bs) {
    __shared__ float sm[KK * 2];
    int t = threadIdx.x;
    {
        int k = t >> 1, d = t & 1;
        float inv = 0.5f + 1.5f * (1.0f / (1.0f + expf(-latent[d])));
        sm[t] = modes[k * 2 + d] * inv;
    }
    __syncthreads();
    int idx = blockIdx.x * 256 + t;
    if (idx >= BB * NN) return;
    int b = idx / NN, n = idx % NN;
    float p0 = nodes[(b * NN + n) * 2 + 0];
    float p1 = nodes[(b * NN + n) * 2 + 1];
    for (int k = 0; k < KK; ++k) {
        float tt = p0 * sm[2 * k] + p1 * sm[2 * k + 1];
        Bc[(b * KK + k) * NN + n] = (__bf16)cosf(tt);
        Bs[(b * KK + k) * NN + n] = (__bf16)sinf(tt);
    }
}

// fc0: h[b][c][n] (channel-major) and ht[b][n][c] (node-major)
__global__ void k_fc0(const float* x, const float* W, const float* bvec,
                      float* h, float* ht) {
    __shared__ float sw[CC * 3 + CC];
    int t = threadIdx.x;
    if (t < CC * 3) sw[t] = W[t];
    if (t < CC) sw[CC * 3 + t] = bvec[t];
    __syncthreads();
    int idx = blockIdx.x * 256 + t;
    if (idx >= BB * NN) return;
    int b = idx / NN, n = idx % NN;
    float x0 = x[idx * 3 + 0], x1 = x[idx * 3 + 1], x2 = x[idx * 3 + 2];
    for (int c = 0; c < CC; ++c) {
        float v = sw[CC * 3 + c] + sw[c * 3] * x0 + sw[c * 3 + 1] * x1 + sw[c * 3 + 2] * x2;
        h[(b * CC + c) * NN + n] = v;
        ht[idx * CC + c] = v;
    }
}

// X0[b][i] = sum_n h[b][i][n] * nw[b][n]
__global__ void k_x0(const float* h, const float* nw, float* X0) {
    int b = blockIdx.x / CC, i = blockIdx.x % CC;
    __shared__ float red[256];
    float s = 0.0f;
    for (int n = threadIdx.x; n < NN; n += 256)
        s += h[(b * CC + i) * NN + n] * nw[b * NN + n];
    red[threadIdx.x] = s;
    __syncthreads();
    for (int st = 128; st > 0; st >>= 1) {
        if (threadIdx.x < st) red[threadIdx.x] += red[threadIdx.x + st];
        __syncthreads();
    }
    if (threadIdx.x == 0) X0[b * CC + i] = red[0];
}

// encode: Xc/Xs[b][i][k] = sum_x (h*nw)[i][x] * B{c,s}[x][k]
// split-K with f32 atomics; K=64 per barrier (2 WMMA per acc per step)
#define ENC_CHUNKS 10
#define ENC_STEPS  (NN / ENC_CHUNKS / 64)   // 25
__global__ void k_encode(const float* h, const float* nw,
                         const __bf16* Bc, const __bf16* Bs,
                         float* Xc, float* Xs) {
    __shared__ __attribute__((aligned(16))) __bf16 As[2 * 2048];
    __shared__ __attribute__((aligned(16))) __bf16 Bpc[2 * 1024];
    __shared__ __attribute__((aligned(16))) __bf16 Bps[2 * 1024];
    int blk = blockIdx.x;
    int b   = blk / (4 * ENC_CHUNKS);
    int r   = blk % (4 * ENC_CHUNKS);
    int kbase = (r / ENC_CHUNKS) * 32;
    int x0    = (r % ENC_CHUNKS) * (NN / ENC_CHUNKS);
    int t = threadIdx.x, lane = t & 31, w = t >> 5;
    int mi = w & 3, nc = w >> 2;   // 4 M-tiles x 2 mode-tiles
    v8f accC = {}; v8f accS = {};
    for (int s = 0; s < ENC_STEPS; ++s) {
        int xb = x0 + s * 64;
        // A: 64 rows x 64 k = (h*nw)->bf16, two frag-order halves
#pragma unroll
        for (int jj = 0; jj < 8; ++jj) {
            int p = t + 256 * jj;                 // 0..2047
            int row = p >> 5, k = (p & 31) << 1;  // k 0..62
            float2 hv = *(const float2*)(h + (b * CC + row) * NN + xb + k);
            float2 wv = *(const float2*)(nw + b * NN + xb + k);
            *(unsigned*)(As + ((k >> 5) << 11) + a_idx(row, k & 31)) =
                pack_bf2(hv.x * wv.x, hv.y * wv.y);
        }
        // B panels: rows = x (K dim), cols = mode; source contiguous along x
#pragma unroll
        for (int jj = 0; jj < 4; ++jj) {
            int p = t + 256 * jj;                 // 0..1023
            int col = p >> 5, k = (p & 31) << 1;
            unsigned vc = *(const unsigned*)(Bc + (kbase + col) * NN + xb + k);
            unsigned vs = *(const unsigned*)(Bs + (kbase + col) * NN + xb + k);
            int bi = ((k >> 5) << 10) + b_idx(k & 31, col);
            *(unsigned*)(Bpc + bi) = vc;
            *(unsigned*)(Bps + bi) = vs;
        }
        if (s + 1 < ENC_STEPS) {   // prefetch next step (global_prefetch_b8)
            int row = t >> 4, k = (t & 15) << 1;
            __builtin_prefetch((const void*)(h + (b * CC + row) * NN + xb + 64 + k), 0, 3);
            __builtin_prefetch((const void*)(Bc + (kbase + (t >> 4)) * NN + xb + 64 + k), 0, 3);
            __builtin_prefetch((const void*)(Bs + (kbase + (t >> 4)) * NN + xb + 64 + k), 0, 3);
        }
        __syncthreads();
#pragma unroll
        for (int half = 0; half < 2; ++half) {
            v16bf a  = frag_ld(As + (half << 11) + ((mi << 5) + lane) * 16);
            v16bf bc = frag_ld(Bpc + (half << 10) + ((nc << 5) + lane) * 16);
            v16bf bs = frag_ld(Bps + (half << 10) + ((nc << 5) + lane) * 16);
            accC = WMMA_BF16(a, bc, accC);
            accS = WMMA_BF16(a, bs, accS);
        }
        __syncthreads();
    }
#pragma unroll
    for (int rr = 0; rr < 8; ++rr) {
        int row = mi * 16 + ((lane >> 4) << 3) + rr;
        int col = kbase + nc * 16 + (lane & 15);
        atomicAdd(&Xc[(b * CC + row) * KK + col], accC[rr]);
        atomicAdd(&Xs[(b * CC + row) * KK + col], accS[rr]);
    }
}

__global__ void k_f0(const float* X0, const float* W0, int l, float* f0) {
    int b = blockIdx.x, o = threadIdx.x;
    float s = 0.0f;
    for (int i = 0; i < CC; ++i) s += X0[b * CC + i] * W0[(l * CC + i) * CC + o];
    f0[b * CC + o] = s;
}

// f_c/f_s mixing + pack decode A-panel: Af[b][o][0:128]=2*f_c, [128:256]=-2*f_s
// block = (b,o), thread = k: all weight/X streams unit-stride
__global__ void k_combine(const float* Xc, const float* Xs,
                          const float* Wc, const float* Ws, int l, __bf16* Af) {
    int b = blockIdx.x / CC, o = blockIdx.x % CC, k = threadIdx.x;
    float fc = 0.0f, fs = 0.0f;
    for (int i = 0; i < CC; ++i) {
        float xc = Xc[(b * CC + i) * KK + k];
        float S  = Xs[(b * CC + i) * KK + k];   // S = -x_s
        float wc = Wc[((l * CC + i) * CC + o) * KK + k];
        float ws = Ws[((l * CC + i) * CC + o) * KK + k];
        fc += xc * wc + S * ws;
        fs += xc * ws - S * wc;
    }
    Af[(b * CC + o) * 256 + k]      = (__bf16)(2.0f * fc);
    Af[(b * CC + o) * 256 + KK + k] = (__bf16)(-2.0f * fs);
}

// gradient field scatter: g[b][n][ch] (node-major, coalesced atomics)
__global__ void k_grad(const int* edges, const float* egw, const float* ht, float* g) {
    unsigned int idx = blockIdx.x * 256u + threadIdx.x;
    if (idx >= (unsigned)(BB * EE) * 128u) return;
    int ch = idx & 127;
    int be = idx >> 7;
    int b = be / EE, e = be % EE;
    int c = ch >> 1, d = ch & 1;
    int tgt = edges[(b * EE + e) * 2 + 0];
    int src = edges[(b * EE + e) * 2 + 1];
    float diff = ht[(b * NN + src) * CC + c] - ht[(b * NN + tgt) * CC + c];
    float wv = egw[(b * EE + e) * 2 + d];
    atomicAdd(&g[(b * NN + tgt) * 128 + ch], wv * diff);
}

// decode: h' = x1 + x2 + x3 (+biases) [+gelu]; K=448 total, K=64 per barrier
__global__ void k_decode(const float* hin, const float* g, const __bf16* Af,
                         const __bf16* wsb, const __bf16* gwsb,
                         const __bf16* Bc, const __bf16* Bs,
                         const float* f0, const float* ws_b, const float* gws_b,
                         int l, int apply_gelu, float* hout, float* ht) {
    __shared__ __attribute__((aligned(16))) __bf16 As[2 * 2048];
    __shared__ __attribute__((aligned(16))) __bf16 Bp[2 * 1024];
    int b  = blockIdx.x / (NN / 32);
    int n0 = (blockIdx.x % (NN / 32)) * 32;
    int t = threadIdx.x, lane = t & 31, w = t >> 5;
    int mi = w & 3, nc = w >> 2;
    int fa = ((mi << 5) + lane) * 16;   // A-frag offset for this wave
    int fb = ((nc << 5) + lane) * 16;   // B-frag offset
    v8f acc = {};
    // spectral: 4 steps; half0 = cos panel x Bc, half1 = sin panel x Bs
    for (int s = 0; s < 4; ++s) {
        int kb = s * 32;
        stage_a_bf16(As,        Af + b * CC * 256 + kb,       256, t, 4);
        stage_a_bf16(As + 2048, Af + b * CC * 256 + 128 + kb, 256, t, 4);
#pragma unroll
        for (int jj = 0; jj < 2; ++jj) {
            int p = t + 256 * jj;
            int col = p >> 4, k = (p & 15) << 1;
            const __bf16* spc = Bc + (kb + k) * NN + n0 + col;
            const __bf16* sps = Bs + (kb + k) * NN + n0 + col;
            BF2 rc; rc.e[0] = spc[0]; rc.e[1] = spc[NN];
            BF2 rs; rs.e[0] = sps[0]; rs.e[1] = sps[NN];
            int bi = b_idx(k, col);
            *(unsigned*)(Bp + bi)        = rc.u;
            *(unsigned*)(Bp + 1024 + bi) = rs.u;
        }
        if (s + 1 < 4) {
            int kbn = (s + 1) * 32;
            __builtin_prefetch((const void*)(Bc + (kbn + ((t & 15) << 1)) * NN + n0), 0, 3);
            __builtin_prefetch((const void*)(Bs + (kbn + ((t & 15) << 1)) * NN + n0), 0, 3);
        }
        __syncthreads();
        v16bf ac = frag_ld(As + fa);
        v16bf bc = frag_ld(Bp + fb);
        acc = WMMA_BF16(ac, bc, acc);
        v16bf an = frag_ld(As + 2048 + fa);
        v16bf bn = frag_ld(Bp + 1024 + fb);
        acc = WMMA_BF16(an, bn, acc);
        __syncthreads();
    }
    // ws_W @ h : one K=64 step
    {
        stage_a_bf16(As,        wsb + l * CC * CC,      CC, t, 4);
        stage_a_bf16(As + 2048, wsb + l * CC * CC + 32, CC, t, 4);
#pragma unroll
        for (int jj = 0; jj < 4; ++jj) {
            int p = t + 256 * jj;                 // 0..1023
            int col = p >> 5, k = (p & 31) << 1;  // k 0..62
            const float* sp = hin + (b * CC + k) * NN + n0 + col;
            *(unsigned*)(Bp + ((k >> 5) << 10) + b_idx(k & 31, col)) = pack_bf2(sp[0], sp[NN]);
        }
        __syncthreads();
#pragma unroll
        for (int half = 0; half < 2; ++half) {
            v16bf a  = frag_ld(As + (half << 11) + fa);
            v16bf bb = frag_ld(Bp + (half << 10) + fb);
            acc = WMMA_BF16(a, bb, acc);
        }
        __syncthreads();
    }
    // gws_W @ grad_field : two K=64 steps (g node-major, float2 loads)
    for (int s = 0; s < 2; ++s) {
        int kb = s * 64;
        stage_a_bf16(As,        gwsb + l * CC * KK + kb,      KK, t, 4);
        stage_a_bf16(As + 2048, gwsb + l * CC * KK + kb + 32, KK, t, 4);
#pragma unroll
        for (int jj = 0; jj < 4; ++jj) {
            int p = t + 256 * jj;
            int col = p >> 5, k = (p & 31) << 1;
            float2 gv = *(const float2*)(g + (b * NN + n0 + col) * 128 + kb + k);
            *(unsigned*)(Bp + ((k >> 5) << 10) + b_idx(k & 31, col)) = pack_bf2(gv.x, gv.y);
        }
        __syncthreads();
#pragma unroll
        for (int half = 0; half < 2; ++half) {
            v16bf a  = frag_ld(As + (half << 11) + fa);
            v16bf bb = frag_ld(Bp + (half << 10) + fb);
            acc = WMMA_BF16(a, bb, acc);
        }
        __syncthreads();
    }
#pragma unroll
    for (int rr = 0; rr < 8; ++rr) {
        int o = mi * 16 + ((lane >> 4) << 3) + rr;
        int n = n0 + nc * 16 + (lane & 15);
        float v = acc[rr] + f0[b * CC + o] + ws_b[l * CC + o] + gws_b[l * CC + o];
        if (apply_gelu) v = gelu_tanh(v);
        hout[(b * CC + o) * NN + n] = v;
        ht[(b * NN + n) * CC + o] = v;
    }
}

// head: out = fc2 @ gelu(fc1 @ h + b1) + b2; fc1 via WMMA, single K=64 step
__global__ void k_head(const float* hin, const __bf16* fc1b, const float* fc1_bias,
                       const float* fc2_W, const float* fc2_b, float* out) {
    __shared__ __attribute__((aligned(16))) __bf16 As[2 * 4096];
    __shared__ __attribute__((aligned(16))) __bf16 Bp[2 * 512];
    __shared__ float hf[128 * 17];
    int b  = blockIdx.x / (NN / 16);
    int n0 = (blockIdx.x % (NN / 16)) * 16;
    int t = threadIdx.x, lane = t & 31, w = t >> 5;
    v8f acc = {};
    {
        stage_a_bf16(As,        fc1b,      CC, t, 8);   // 128 rows x k 0..31
        stage_a_bf16(As + 4096, fc1b + 32, CC, t, 8);   // 128 rows x k 32..63
#pragma unroll
        for (int jj = 0; jj < 2; ++jj) {
            int p = t + 256 * jj;                 // 0..511
            int col = p >> 5, k = (p & 31) << 1;  // col 0..15, k 0..62
            const float* sp = hin + (b * CC + k) * NN + n0 + col;
            *(unsigned*)(Bp + ((k >> 5) << 9) + b_idx(k & 31, col)) = pack_bf2(sp[0], sp[NN]);
        }
        __syncthreads();
#pragma unroll
        for (int half = 0; half < 2; ++half) {
            v16bf a  = frag_ld(As + (half << 12) + ((w << 5) + lane) * 16);
            v16bf bb = frag_ld(Bp + (half << 9) + lane * 16);
            acc = WMMA_BF16(a, bb, acc);
        }
    }
#pragma unroll
    for (int rr = 0; rr < 8; ++rr) {
        int o = w * 16 + ((lane >> 4) << 3) + rr;
        int n = lane & 15;
        hf[o * 17 + n] = gelu_tanh(acc[rr] + fc1_bias[o]);
    }
    __syncthreads();
    if (t < 16) {
        float s = fc2_b[0];
        for (int o = 0; o < 128; ++o) s += fc2_W[o] * hf[o * 17 + t];
        out[b * NN + n0 + t] = s;
    }
}

// ------------------------------------------------------------------
extern "C" void kernel_launch(void* const* d_in, const int* in_sizes, int n_in,
                              void* d_out, int out_size, void* d_ws, size_t ws_size,
                              hipStream_t stream) {
    (void)in_sizes; (void)n_in; (void)out_size; (void)ws_size;
    const float* x      = (const float*)d_in[0];
    const float* nodes  = (const float*)d_in[2];
    const float* nw     = (const float*)d_in[3];
    const int*   edges  = (const int*)d_in[4];
    const float* egw    = (const float*)d_in[5];
    const float* modes  = (const float*)d_in[6];
    const float* latent = (const float*)d_in[7];
    const float* fc0_W  = (const float*)d_in[8];
    const float* fc0_b  = (const float*)d_in[9];
    const float* Wc     = (const float*)d_in[10];
    const float* Ws     = (const float*)d_in[11];
    const float* W0     = (const float*)d_in[12];
    const float* ws_W   = (const float*)d_in[13];
    const float* ws_b   = (const float*)d_in[14];
    const float* gws_W  = (const float*)d_in[15];
    const float* gws_b  = (const float*)d_in[16];
    const float* fc1_W  = (const float*)d_in[17];
    const float* fc1_b  = (const float*)d_in[18];
    const float* fc2_W  = (const float*)d_in[19];
    const float* fc2_b  = (const float*)d_in[20];
    float* outp = (float*)d_out;

    char* base = (char*)d_ws;
    size_t off = 0;
    auto alloc = [&](size_t bytes) -> void* {
        void* p = base + off;
        off = (off + bytes + 255) & ~(size_t)255;
        return p;
    };
    __bf16* Bc   = (__bf16*)alloc((size_t)BB * KK * NN * 2);
    __bf16* Bs   = (__bf16*)alloc((size_t)BB * KK * NN * 2);
    float*  h0   = (float*) alloc((size_t)BB * CC * NN * 4);
    float*  h1   = (float*) alloc((size_t)BB * CC * NN * 4);
    float*  ht   = (float*) alloc((size_t)BB * NN * CC * 4);
    float*  g    = (float*) alloc((size_t)BB * NN * 128 * 4);
    float*  Xcs  = (float*) alloc((size_t)2 * BB * CC * KK * 4);  // Xc | Xs adjacent
    float*  Xc   = Xcs;
    float*  Xs   = Xcs + (size_t)BB * CC * KK;
    float*  X0   = (float*) alloc((size_t)BB * CC * 4);
    float*  f0   = (float*) alloc((size_t)BB * CC * 4);
    __bf16* Af   = (__bf16*)alloc((size_t)BB * CC * 256 * 2);
    __bf16* wsb  = (__bf16*)alloc((size_t)LL * CC * CC * 2);
    __bf16* gwsb = (__bf16*)alloc((size_t)LL * CC * KK * 2);
    __bf16* fc1bw= (__bf16*)alloc((size_t)KK * CC * 2);

    const int prep_n = LL * CC * CC + LL * CC * KK + KK * CC;
    k_prep<<<(prep_n + 255) / 256, 256, 0, stream>>>(ws_W, gws_W, fc1_W, wsb, gwsb, fc1bw);
    k_bases<<<(BB * NN + 255) / 256, 256, 0, stream>>>(nodes, modes, latent, Bc, Bs);
    k_fc0<<<(BB * NN + 255) / 256, 256, 0, stream>>>(x, fc0_W, fc0_b, h0, ht);

    float* hcur = h0;
    float* hnxt = h1;
    for (int l = 0; l < LL; ++l) {
        const int nXcs4 = (2 * BB * CC * KK) / 4;
        k_zero4<<<(nXcs4 + 255) / 256, 256, 0, stream>>>((float4*)Xcs, nXcs4);
        const int nG4 = (BB * NN * 128) / 4;
        k_zero4<<<(nG4 + 255) / 256, 256, 0, stream>>>((float4*)g, nG4);

        k_x0<<<BB * CC, 256, 0, stream>>>(hcur, nw, X0);
        k_encode<<<BB * 4 * ENC_CHUNKS, 256, 0, stream>>>(hcur, nw, Bc, Bs, Xc, Xs);
        k_f0<<<BB, CC, 0, stream>>>(X0, W0, l, f0);
        k_combine<<<BB * CC, KK, 0, stream>>>(Xc, Xs, Wc, Ws, l, Af);

        const unsigned nGr = (unsigned)(BB * EE) * 128u;
        k_grad<<<(nGr + 255u) / 256u, 256, 0, stream>>>(edges, egw, ht, g);

        k_decode<<<BB * (NN / 32), 256, 0, stream>>>(hcur, g, Af, wsb, gwsb, Bc, Bs,
                                                     f0, ws_b, gws_b, l,
                                                     (l < LL - 1) ? 1 : 0, hnxt, ht);
        float* tmp = hcur; hcur = hnxt; hnxt = tmp;
    }

    k_head<<<BB * (NN / 16), 256, 0, stream>>>(hcur, fc1bw, fc1_b, fc2_W, fc2_b, outp);
}
